// TKGraphormer_14955076125441
// MI455X (gfx1250) — matrix operand order repaired and checked by reading, not compiled, not run on hardware
//
#include <hip/hip_runtime.h>
#include <hip/hip_bf16.h>

typedef __attribute__((ext_vector_type(16))) _Float16 v16h;
typedef __attribute__((ext_vector_type(8)))  _Float16 v8h;
typedef __attribute__((ext_vector_type(8)))  float    v8f;
typedef __attribute__((ext_vector_type(4)))  float    vf4;

// D=200 padded to 224 (7 k-blocks of 32); 2D=400 padded to 416 (13 k-blocks).
#define DP  224
#define KP2 416

// ------------------------------------------------------------------
// Utility kernels
// ------------------------------------------------------------------
__global__ void zero_f32_kernel(float* p, long n) {
  long i = (long)blockIdx.x * blockDim.x + threadIdx.x;
  long st = (long)gridDim.x * blockDim.x;
  for (; i < n; i += st) p[i] = 0.f;
}

// Strided row copy: out[r*ldOut + c] = in[r*ldIn + c], c < D
__global__ void copy_rows_kernel(const float* in, int ldIn, float* out, int ldOut,
                                 int rows, int D) {
  int i = blockIdx.x * blockDim.x + threadIdx.x;
  if (i >= rows * D) return;
  int r = i / D, c = i % D;
  out[(long)r * ldOut + c] = in[(long)r * ldIn + c];
}

// Pack fp32 row-major B[K,N] into WMMA B-fragment order (f16, zero-padded):
// P[((tile*nkb + kb)*512) + lane*16 + i] = B[kb*32 + ((lane<16)?0:8) + (i&7) + ((i<8)?0:16), tile*16 + (lane&15)]
__global__ void pack_b_kernel(const float* B, _Float16* P, int K, int N) {
  int nkb = (K + 31) >> 5;
  int ntl = (N + 15) >> 4;
  long total = (long)ntl * nkb * 512;
  long idx = (long)blockIdx.x * blockDim.x + threadIdx.x;
  long st = (long)gridDim.x * blockDim.x;
  for (; idx < total; idx += st) {
    int i = (int)(idx & 15);
    int lane = (int)((idx >> 4) & 31);
    long fb = idx >> 9;
    int kb = (int)(fb % nkb);
    int t = (int)(fb / nkb);
    int n = t * 16 + (lane & 15);
    int k = kb * 32 + ((lane < 16) ? 0 : 8) + (i & 7) + ((i < 8) ? 0 : 16);
    P[idx] = (k < K && n < N) ? (_Float16)B[(long)k * N + n] : (_Float16)0.f;
  }
}

// One block per row; D <= 256. Strided in/out. Safe in-place.
__global__ void l2norm_rows_kernel(const float* in, int ldIn, float* out, int ldOut, int D) {
  int row = blockIdx.x;
  __shared__ float red[256];
  int c = threadIdx.x;
  float v = 0.f, s = 0.f;
  if (c < D) { v = in[(long)row * ldIn + c]; s = v * v; }
  red[threadIdx.x] = s;
  __syncthreads();
  for (int off = 128; off > 0; off >>= 1) {
    if (threadIdx.x < off) red[threadIdx.x] += red[threadIdx.x + off];
    __syncthreads();
  }
  float n = fmaxf(sqrtf(red[0]), 1e-12f);
  if (c < D) out[(long)row * ldOut + c] = v / n;
}

// One wave per r2e entry: scatter ent_embeds row into per-relation sum + count.
__global__ void r2e_scatter_kernel(const float* ent, const int* r2e_ent, const int* r2e_rel,
                                   float* ssum, float* cnt, int M, int D) {
  int idx = blockIdx.x * (blockDim.x >> 5) + (threadIdx.x >> 5);
  if (idx >= M) return;
  int lane = threadIdx.x & 31;
  int e = r2e_ent[idx];
  int r = r2e_rel[idx];
  const float* src = ent + (long)e * D;
  float* dstp = ssum + (long)r * D;
  for (int c = lane; c < D; c += 32) atomicAdd(&dstp[c], src[c]);
  if (lane == 0) atomicAdd(&cnt[r], 1.0f);
}

// gi = x_input @ w_ih^T + b_ih ; gh = h0 @ w_hh^T + b_hh   (n_rel x 3D outputs)
__global__ void gru_gates_kernel(const float* rel_embeds, const float* ssum, const float* cnt,
                                 const float* h0, const float* w_ih, const float* w_hh,
                                 const float* b_ih, const float* b_hh,
                                 float* gi, float* gh, int n_rel, int D) {
  int j = blockIdx.x * blockDim.x + threadIdx.x;
  if (j >= n_rel * 3 * D) return;
  int r = j / (3 * D);
  int g = j % (3 * D);
  float cinv = 1.0f / fmaxf(cnt[r], 1.0f);
  const float* wi = w_ih + (long)g * (2 * D);
  const float* wh = w_hh + (long)g * D;
  const float* re = rel_embeds + (long)r * D;
  const float* ss = ssum + (long)r * D;
  const float* hr = h0 + (long)r * DP;
  float ai = b_ih[g], ah = b_hh[g];
  for (int k = 0; k < D; ++k) ai += re[k] * wi[k];
  for (int k = 0; k < D; ++k) ai += (ss[k] * cinv) * wi[D + k];
  for (int k = 0; k < D; ++k) ah += hr[k] * wh[k];
  gi[j] = ai;
  gh[j] = ah;
}

// GRU elementwise combine, in-place on h0 (stride DP).
__global__ void gru_combine_kernel(const float* gi, const float* gh, float* h0, int n_rel, int D) {
  int j = blockIdx.x * blockDim.x + threadIdx.x;
  if (j >= n_rel * D) return;
  int r = j / D, c = j % D;
  const float* gir = gi + (long)r * 3 * D;
  const float* ghr = gh + (long)r * 3 * D;
  float rg = 1.f / (1.f + expf(-(gir[c] + ghr[c])));
  float z  = 1.f / (1.f + expf(-(gir[D + c] + ghr[D + c])));
  float nn = tanhf(gir[2 * D + c] + rg * ghr[2 * D + c]);
  long o = (long)r * DP + c;
  h0[o] = (1.f - z) * nn + z * h0[o];
}

__global__ void deg_count_kernel(const int* dst, float* deg, int E) {
  int e = blockIdx.x * blockDim.x + threadIdx.x;
  if (e < E) atomicAdd(&deg[dst[e]], 1.0f);
}

// Fused: cur = relu(agg/deg + loopO); l2norm(cur); tw = sigmoid(gateO + bias);
// h_next = tw*cur + (1-tw)*h.  One block per entity row. All strides DP.
__global__ void entity_update_kernel(const float* h, const float* agg, const float* deg,
                                     const float* loopO, const float* gateO, const float* tg_bias,
                                     float* h_next, int D) {
  int row = blockIdx.x;
  __shared__ float red[256];
  int c = threadIdx.x;
  float dg = fmaxf(deg[row], 1.0f);
  float cur = 0.f, s = 0.f;
  long o = (long)row * DP + c;
  if (c < D) {
    cur = fmaxf(agg[o] / dg + loopO[o], 0.f);
    s = cur * cur;
  }
  red[threadIdx.x] = s;
  __syncthreads();
  for (int off = 128; off > 0; off >>= 1) {
    if (threadIdx.x < off) red[threadIdx.x] += red[threadIdx.x + off];
    __syncthreads();
  }
  float nrm = fmaxf(sqrtf(red[0]), 1e-12f);
  if (c < D) {
    float hv = h[o];
    float tw = 1.f / (1.f + expf(-(gateO[o] + tg_bias[c])));
    h_next[o] = tw * (cur / nrm) + (1.f - tw) * hv;
  }
}

// X[b, 0:D] = h[qe[b]]; X[b, D:2D] = h0[qr[b]]; X cols [2D,KP2) = 0.  (f16, ld=KP2)
__global__ void build_X_kernel(const float* h, const float* h0, const int* qe, const int* qr,
                               _Float16* X, int B, int D) {
  int i = blockIdx.x * blockDim.x + threadIdx.x;
  if (i >= B * KP2) return;
  int b = i / KP2, c = i % KP2;
  float v = 0.f;
  if (c < D) v = h[(long)qe[b] * DP + c];
  else if (c < 2 * D) v = h0[(long)qr[b] * DP + (c - D)];
  X[i] = (_Float16)v;
}

// ------------------------------------------------------------------
// WMMA fragment loads (CDNA5 ISA 7.12.2 layouts). All unconditional.
// A lane l: M=l&15 ; halves 0..7 -> K = kb + ((l<16)?0:8) + i ; halves 8..15 -> +16
// B packed contiguously per (tile, kblock, lane).
// D/C lane l: N=l&15 ; acc[v] -> M = v + ((l<16)?0:8)
// ------------------------------------------------------------------
__device__ __forceinline__ v16h cvt_a_f32(const float* ap) {
  vf4 x0 = *(const vf4*)(ap);
  vf4 x1 = *(const vf4*)(ap + 4);
  vf4 x2 = *(const vf4*)(ap + 16);
  vf4 x3 = *(const vf4*)(ap + 20);
  v16h a;
#pragma unroll
  for (int i = 0; i < 4; ++i) {
    a[i]      = (_Float16)x0[i];
    a[i + 4]  = (_Float16)x1[i];
    a[i + 8]  = (_Float16)x2[i];
    a[i + 12] = (_Float16)x3[i];
  }
  return a;
}

__device__ __forceinline__ v16h load_a_f16(const _Float16* ap) {
  v8h lo = *(const v8h*)(ap);
  v8h hi = *(const v8h*)(ap + 16);
  v16h a;
#pragma unroll
  for (int i = 0; i < 8; ++i) { a[i] = lo[i]; a[i + 8] = hi[i]; }
  return a;
}

// Generic GEMM: Out[M,N] = act(A[M,*ldA] @ Bpacked + bias)
// A fp32 (cvt at load) or f16; Out fp32 or f16 with stride ldOut.
// K handled via nkb packed blocks; A stride must be >= nkb*32 (garbage pad ok,
// since B pad region is exact zero).
template <bool A_IS_HALF, bool RELU, bool HAS_BIAS, bool OUT_HALF>
__global__ void wmma_gemm_kernel(const void* Av, int ldA, const _Float16* Bp,
                                 const float* bias, void* Outv, int ldOut,
                                 int M, int N, int nkb) {
  int wid = blockIdx.x * (blockDim.x >> 5) + (threadIdx.x >> 5);
  int ntiles = (N + 15) >> 4;
  int mtiles = (M + 15) >> 4;
  bool valid = wid < ntiles * mtiles;
  int w = valid ? wid : 0;
  int mrow = (w / ntiles) * 16;
  int nt = w % ntiles;
  int ncol = nt * 16;
  int lane = threadIdx.x & 31;

  int row = mrow + (lane & 15);
  int rowc = (row < M) ? row : (M - 1);
  int kA = (lane < 16) ? 0 : 8;

  const float* af = (const float*)Av + (A_IS_HALF ? 0 : ((long)rowc * ldA + kA));
  const _Float16* ah = (const _Float16*)Av + (A_IS_HALF ? ((long)rowc * ldA + kA) : 0);
  const _Float16* bp = Bp + (long)nt * nkb * 512 + lane * 16;

  v8f acc = {};
  for (int kb = 0; kb < nkb; ++kb) {
    v16h a = A_IS_HALF ? load_a_f16(ah) : cvt_a_f32(af);
    v16h b = *(const v16h*)bp;
    acc = __builtin_amdgcn_wmma_f32_16x16x32_f16(false, a, false, b, (short)0, acc,
                                                 false, false);
    af += 32; ah += 32; bp += 512;
  }

  int col = ncol + (lane & 15);
#pragma unroll
  for (int v = 0; v < 8; ++v) {
    int r = mrow + v + ((lane < 16) ? 0 : 8);
    if (valid && r < M && col < N) {
      float val = acc[v];
      if (HAS_BIAS) val += bias[col];
      if (RELU) val = fmaxf(val, 0.f);
      if (OUT_HALF) ((_Float16*)Outv)[(long)r * ldOut + col] = (_Float16)val;
      else          ((float*)Outv)[(long)r * ldOut + col] = val;
    }
  }
}

// Edge message GEMM with gathered A rows + atomic scatter:
// msg[e,:] = (h[src[e]] + h0[etype[e]]) @ Wmsg ; agg[dst[e],:] += msg[e,:]
// h/h0/agg strides = DP; Wmsg packed (nkb = DP/32 = 7).
__global__ void edge_msg_wmma_kernel(const float* h, const float* h0, const _Float16* Bp,
                                     const int* src, const int* etype, const int* dst,
                                     float* agg, int E, int D) {
  int wid = blockIdx.x * (blockDim.x >> 5) + (threadIdx.x >> 5);
  const int nkb = DP / 32;
  int ntiles = (D + 15) >> 4;
  int etiles = (E + 15) >> 4;
  bool valid = wid < ntiles * etiles;
  int w = valid ? wid : 0;
  int ebase = (w / ntiles) * 16;
  int nt = w % ntiles;
  int ncol = nt * 16;
  int lane = threadIdx.x & 31;

  int e = ebase + (lane & 15);
  int ec = (e < E) ? e : (E - 1);
  int kA = (lane < 16) ? 0 : 8;
  const float* hp = h + (long)src[ec] * DP + kA;
  const float* rp = h0 + (long)etype[ec] * DP + kA;
  const _Float16* bp = Bp + (long)nt * nkb * 512 + lane * 16;

  v8f acc = {};
  for (int kb = 0; kb < nkb; ++kb) {
    vf4 h0v = *(const vf4*)(hp)      + *(const vf4*)(rp);
    vf4 h1v = *(const vf4*)(hp + 4)  + *(const vf4*)(rp + 4);
    vf4 h2v = *(const vf4*)(hp + 16) + *(const vf4*)(rp + 16);
    vf4 h3v = *(const vf4*)(hp + 20) + *(const vf4*)(rp + 20);
    v16h a;
#pragma unroll
    for (int i = 0; i < 4; ++i) {
      a[i]      = (_Float16)h0v[i];
      a[i + 4]  = (_Float16)h1v[i];
      a[i + 8]  = (_Float16)h2v[i];
      a[i + 12] = (_Float16)h3v[i];
    }
    v16h b = *(const v16h*)bp;
    acc = __builtin_amdgcn_wmma_f32_16x16x32_f16(false, a, false, b, (short)0, acc,
                                                 false, false);
    hp += 32; rp += 32; bp += 512;
  }

  int col = ncol + (lane & 15);
#pragma unroll
  for (int v = 0; v < 8; ++v) {
    int er = ebase + v + ((lane < 16) ? 0 : 8);
    if (valid && er < E && col < D) {
      int dd = dst[er];
      atomicAdd(&agg[(long)dd * DP + col], acc[v]);
    }
  }
}

// ------------------------------------------------------------------
// Launch
// ------------------------------------------------------------------
extern "C" void kernel_launch(void* const* d_in, const int* in_sizes, int n_in,
                              void* d_out, int out_size, void* d_ws, size_t ws_size,
                              hipStream_t stream) {
  const int N_ENT = 100000, N_REL = 230, D = 200, T = 3, E = 200000, Mr = 400000, Bq = 1024;

  const float* ent_embeds = (const float*)d_in[0];
  const float* rel_embeds = (const float*)d_in[1];
  const float* tg_w   = (const float*)d_in[2];
  const float* tg_b   = (const float*)d_in[3];
  const float* w_ih   = (const float*)d_in[4];
  const float* w_hh   = (const float*)d_in[5];
  const float* b_ih   = (const float*)d_in[6];
  const float* b_hh   = (const float*)d_in[7];
  const float* w_msg  = (const float*)d_in[8];
  const float* w_loop = (const float*)d_in[9];
  const float* dec_w1 = (const float*)d_in[10];
  const float* dec_b1 = (const float*)d_in[11];
  const float* dec_w2 = (const float*)d_in[12];
  const float* dec_b2 = (const float*)d_in[13];
  const int* src   = (const int*)d_in[14];
  const int* dst   = (const int*)d_in[15];
  const int* etype = (const int*)d_in[16];
  const int* r2e_ent = (const int*)d_in[17];
  const int* r2e_rel = (const int*)d_in[18];
  const int* qe = (const int*)d_in[19];
  const int* qr = (const int*)d_in[20];

  unsigned char* ws = (unsigned char*)d_ws;
  size_t off = 0;
  auto alloc = [&](size_t bytes) -> void* {
    void* p = (void*)(ws + off);
    off += (bytes + 255) & ~(size_t)255;
    return p;
  };

  const size_t entBuf = (size_t)N_ENT * DP * 4;         // 89.6 MB
  float* hA   = (float*)alloc(entBuf);
  float* hB   = (float*)alloc(entBuf);
  float* agg  = (float*)alloc(entBuf);
  void*  big1 = alloc(entBuf);   // loopO (f32) -> packed dec_w2 f16 (83.2 MB)
  void*  big2 = alloc(entBuf);   // gateO (f32) -> X/hid f16
  float* deg  = (float*)alloc((size_t)N_ENT * 4);
  float* ssum = (float*)alloc((size_t)N_REL * D * 4);
  float* cnt  = (float*)alloc((size_t)N_REL * 4);
  float* h0   = (float*)alloc((size_t)N_REL * DP * 4);
  float* gi   = (float*)alloc((size_t)N_REL * 3 * D * 4);
  float* gh   = (float*)alloc((size_t)N_REL * 3 * D * 4);
  // Packed weight fragments: (N/16) tiles * (Kpad/32) kblocks * 1024B
  _Float16* wmsgP  = (_Float16*)alloc((size_t)13 * 7 * 1024);
  _Float16* wloopP = (_Float16*)alloc((size_t)13 * 7 * 1024);
  _Float16* wgateP = (_Float16*)alloc((size_t)13 * 7 * 1024);
  _Float16* w1P    = (_Float16*)alloc((size_t)25 * 13 * 1024);

  float* loopO = (float*)big1;
  float* gateO = (float*)big2;

  dim3 blk(256);

  // 0) Pack weights into fragment order (zero-padded K).
  pack_b_kernel<<<dim3(128), blk, 0, stream>>>(w_msg, wmsgP, D, D);
  pack_b_kernel<<<dim3(128), blk, 0, stream>>>(w_loop, wloopP, D, D);
  pack_b_kernel<<<dim3(128), blk, 0, stream>>>(tg_w, wgateP, D, D);
  pack_b_kernel<<<dim3(512), blk, 0, stream>>>(dec_w1, w1P, 2 * D, 2 * D);

  // 1) Zero padded state buffers once (so K-pad columns are exact zeros), then
  //    h = l2norm(ent_embeds); h0 = rel_embeds (raw strided copy, per reference).
  zero_f32_kernel<<<dim3(4096), blk, 0, stream>>>(hA, (long)N_ENT * DP);
  zero_f32_kernel<<<dim3(4096), blk, 0, stream>>>(hB, (long)N_ENT * DP);
  zero_f32_kernel<<<dim3(64), blk, 0, stream>>>(h0, (long)N_REL * DP);
  l2norm_rows_kernel<<<dim3(N_ENT), blk, 0, stream>>>(ent_embeds, D, hA, DP, D);
  copy_rows_kernel<<<dim3((N_REL * D + 255) / 256), blk, 0, stream>>>(
      rel_embeds, D, h0, DP, N_REL, D);

  float* hCur = hA;
  float* hNxt = hB;

  for (int t = 0; t < T; ++t) {
    const int* src_t = src + (long)t * E;
    const int* dst_t = dst + (long)t * E;
    const int* ety_t = etype + (long)t * E;
    const int* r2e_e = r2e_ent + (long)t * Mr;
    const int* r2e_r = r2e_rel + (long)t * Mr;

    // relation mean via atomics
    zero_f32_kernel<<<dim3(180), blk, 0, stream>>>(ssum, (long)N_REL * D);
    zero_f32_kernel<<<dim3(1), blk, 0, stream>>>(cnt, (long)N_REL);
    r2e_scatter_kernel<<<dim3((Mr + 7) / 8), blk, 0, stream>>>(ent_embeds, r2e_e, r2e_r,
                                                               ssum, cnt, Mr, D);
    // GRU update of h0 + l2norm (h0 stride DP)
    gru_gates_kernel<<<dim3((N_REL * 3 * D + 255) / 256), blk, 0, stream>>>(
        rel_embeds, ssum, cnt, h0, w_ih, w_hh, b_ih, b_hh, gi, gh, N_REL, D);
    gru_combine_kernel<<<dim3((N_REL * D + 255) / 256), blk, 0, stream>>>(gi, gh, h0, N_REL, D);
    l2norm_rows_kernel<<<dim3(N_REL), blk, 0, stream>>>(h0, DP, h0, DP, D);

    // edge messages -> agg (WMMA + atomic scatter), degree counts
    zero_f32_kernel<<<dim3(4096), blk, 0, stream>>>(agg, (long)N_ENT * DP);
    zero_f32_kernel<<<dim3(256), blk, 0, stream>>>(deg, (long)N_ENT);
    deg_count_kernel<<<dim3((E + 255) / 256), blk, 0, stream>>>(dst_t, deg, E);
    {
      int tiles = ((E + 15) / 16) * ((D + 15) / 16);
      edge_msg_wmma_kernel<<<dim3((tiles + 7) / 8), blk, 0, stream>>>(
          hCur, h0, wmsgP, src_t, ety_t, dst_t, agg, E, D);
    }

    // h @ W_loop, h @ W_gate  (WMMA)
    {
      int tiles = ((N_ENT + 15) / 16) * ((D + 15) / 16);
      wmma_gemm_kernel<false, false, false, false><<<dim3((tiles + 7) / 8), blk, 0, stream>>>(
          hCur, DP, wloopP, nullptr, loopO, DP, N_ENT, D, DP / 32);
      wmma_gemm_kernel<false, false, false, false><<<dim3((tiles + 7) / 8), blk, 0, stream>>>(
          hCur, DP, wgateP, nullptr, gateO, DP, N_ENT, D, DP / 32);
    }

    // fused normalize + time gate
    entity_update_kernel<<<dim3(N_ENT), blk, 0, stream>>>(hCur, agg, deg, loopO, gateO,
                                                          tg_b, hNxt, D);
    float* tmp = hCur; hCur = hNxt; hNxt = tmp;
  }

  // 2) Decoder. Reuse big regions (loopO/gateO dead now).
  _Float16* w2P = (_Float16*)big1;                    // packed [6250][13][512] f16, 83.2 MB
  _Float16* X   = (_Float16*)big2;                    // [Bq, KP2]
  _Float16* hid = X + (size_t)Bq * KP2;               // [Bq, KP2]
  pack_b_kernel<<<dim3(8192), blk, 0, stream>>>(dec_w2, w2P, 2 * D, N_ENT);
  // Zero Bq*KP2 f32 = exactly the X and hid f16 buffers (pad columns become 0).
  zero_f32_kernel<<<dim3(512), blk, 0, stream>>>((float*)X, (long)Bq * KP2);
  build_X_kernel<<<dim3((Bq * KP2 + 255) / 256), blk, 0, stream>>>(hCur, h0, qe, qr, X, Bq, D);

  {
    int tiles = ((Bq + 15) / 16) * ((2 * D + 15) / 16);
    wmma_gemm_kernel<true, true, true, true><<<dim3((tiles + 7) / 8), blk, 0, stream>>>(
        X, KP2, w1P, dec_b1, hid, KP2, Bq, 2 * D, KP2 / 32);
  }
  {
    int tiles = ((Bq + 15) / 16) * ((N_ENT + 15) / 16);
    wmma_gemm_kernel<true, false, true, false><<<dim3((tiles + 7) / 8), blk, 0, stream>>>(
        hid, KP2, w2P, dec_b2, d_out, N_ENT, Bq, N_ENT, KP2 / 32);
  }
}
